// MultilayerGRU_63127429316881
// MI455X (gfx1250) — compile-verified, matrix operand order checked by reading, not run on hardware
//
#include <hip/hip_runtime.h>
#include <hip/hip_bf16.h>

// ---------------------------------------------------------------------------
// Multilayer GRU for MI455X (gfx1250, wave32).
// bf16 WMMA (v_wmma_f32_16x16x32_bf16) with f32 accumulate for all GEMMs;
// f32 master hidden state; async global->LDS weight-tile staging (ASYNCcnt)
// in the big GEMM; global_prefetch for lookahead.
// ---------------------------------------------------------------------------

typedef __bf16 bf16_t;
typedef __attribute__((ext_vector_type(16))) __bf16 v16bf;
typedef __attribute__((ext_vector_type(8)))  __bf16 v8bf;
typedef __attribute__((ext_vector_type(8)))  float  v8f;

constexpr int Bb   = 32;
constexpr int Ss   = 512;
constexpr int Hh   = 1024;
constexpr int Ll   = 3;
constexpr int Oo   = 1024;
constexpr int MTOT = Bb * Ss;     // 16384

// ---------------------------------------------------------------------------
// Device helpers
// ---------------------------------------------------------------------------
__device__ inline v8f vzero8() {
  v8f z;
#pragma unroll
  for (int i = 0; i < 8; ++i) z[i] = 0.0f;
  return z;
}

// Build a 16x32 (M x K) bf16 WMMA fragment row for this lane.
// ISA layout: lane<16 holds K in {0..7} u {16..23}; lane>=16 holds {8..15} u {24..31}.
// base points at (row, k0); half = lane>>4.
__device__ inline v16bf load_frag(const bf16_t* __restrict__ base, int half) {
  const v8bf lo = *reinterpret_cast<const v8bf*>(base + half * 8);
  const v8bf hi = *reinterpret_cast<const v8bf*>(base + 16 + half * 8);
  v16bf f;
#pragma unroll
  for (int i = 0; i < 8; ++i) { f[i] = lo[i]; f[8 + i] = hi[i]; }
  return f;
}

__device__ inline float sigmoid_f(float x) {
  return 1.0f / (1.0f + __expf(-x));
}

__device__ inline float tanh_f(float x) {
  x = fminf(fmaxf(x, -15.0f), 15.0f);
  float e = __expf(2.0f * x);
  return (e - 1.0f) / (e + 1.0f);
}

// ---------------------------------------------------------------------------
// f32 -> bf16 conversion
// ---------------------------------------------------------------------------
__global__ void cvt_f32_bf16(const float* __restrict__ in, bf16_t* __restrict__ out,
                             size_t n) {
  size_t i = (size_t)blockIdx.x * blockDim.x + threadIdx.x;
  if (i < n) out[i] = (bf16_t)in[i];
}

// ---------------------------------------------------------------------------
// C[M,N] = A[M,K] @ W[N,K]^T + bias[N]
// A,W bf16 row-major; C f32. Block: 256 thr (8 waves). Tile M=128,N=64,Kstep=32.
// Wave w owns M-subtile w; loops 4 N-subtiles. W tile staged via the gfx1250
// async global->LDS path (no VGPR round-trip; completion on ASYNCcnt).
// grid = (N/64, M/128)
// ---------------------------------------------------------------------------
__global__ __launch_bounds__(256)
void gemm_bias_kernel(const bf16_t* __restrict__ A, const bf16_t* __restrict__ W,
                      const float* __restrict__ bias, float* __restrict__ C,
                      int M, int N, int K) {
  __shared__ __align__(16) bf16_t sW[64 * 32];

  const int lane  = threadIdx.x & 31;
  const int wave  = threadIdx.x >> 5;
  const int r     = lane & 15;
  const int half  = lane >> 4;
  const int mBase = blockIdx.y * 128 + wave * 16;
  const int nBase = blockIdx.x * 64;

  v8f acc[4];
#pragma unroll
  for (int i = 0; i < 4; ++i) acc[i] = vzero8();

  const bf16_t* Arow = A + (size_t)(mBase + r) * K;
  const int trow = threadIdx.x >> 2;   // 0..63
  const int tch  = threadIdx.x & 3;    // 0..3
  const bf16_t* Wld = W + (size_t)(nBase + trow) * K + tch * 8;

  // LDS byte offset of this thread's 16B staging slot (generic shared address
  // truncated to 32 bits == LDS offset per ISA flat-address mapping).
  const unsigned lds_dst = (unsigned)(uintptr_t)(&sW[trow * 32 + tch * 8]);

  for (int k0 = 0; k0 < K; k0 += 32) {
    // async stage of W tile [64 x 32] into LDS: one b128 per thread,
    // global -> LDS directly, tracked by ASYNCcnt.
    {
      const unsigned long long gsrc = (unsigned long long)(uintptr_t)(Wld + k0);
      asm volatile("global_load_async_to_lds_b128 %0, %1, off"
                   :: "v"(lds_dst), "v"(gsrc) : "memory");
    }
    if (k0 + 32 < K) {
      __builtin_prefetch(Wld + k0 + 32, 0, 1);   // global_prefetch_b8
      __builtin_prefetch(Arow + k0 + 32, 0, 1);
    }
    asm volatile("s_wait_asynccnt 0" ::: "memory");
    __syncthreads();

    const v16bf a = load_frag(Arow + k0, half);
#pragma unroll
    for (int ns = 0; ns < 4; ++ns) {
      const v16bf b = load_frag(&sW[(ns * 16 + r) * 32], half);
      acc[ns] = __builtin_amdgcn_wmma_f32_16x16x32_bf16(
          false, a, false, b, (short)0, acc[ns], false, false);
    }
    __syncthreads();
  }

#pragma unroll
  for (int ns = 0; ns < 4; ++ns) {
    const int n  = nBase + ns * 16 + r;
    const float bv = bias ? bias[n] : 0.0f;
#pragma unroll
    for (int v = 0; v < 8; ++v) {
      const int m = mBase + (half ? v + 8 : v);
      C[(size_t)m * N + n] = acc[ns][v] + bv;
    }
  }
}

// ---------------------------------------------------------------------------
// h init from h0[:, l, :]
// ---------------------------------------------------------------------------
__global__ void init_h_kernel(const float* __restrict__ h0, int layer,
                              float* __restrict__ hf, bf16_t* __restrict__ hb) {
  int i = blockIdx.x * blockDim.x + threadIdx.x;   // 0 .. 32*1024-1
  if (i < Bb * Hh) {
    int b = i / Hh, n = i % Hh;
    float v = h0[((size_t)b * Ll + layer) * Hh + n];
    hf[i] = v;
    hb[i] = (bf16_t)v;
  }
}

// ---------------------------------------------------------------------------
// Step A: z = sigmoid(Xz[t] + h @ Wzh^T) ; r = sigmoid(Xr[t] + h @ Wrh^T)
// stores z (f32) and bf16(r*h).
// grid = 16 blocks x 256 thr; wave w: m-tile = w&1, n-subtile = w>>1.
// ---------------------------------------------------------------------------
__global__ __launch_bounds__(256)
void gru_stepA_kernel(const bf16_t* __restrict__ hb, const float* __restrict__ hf,
                      const bf16_t* __restrict__ Wzh, const bf16_t* __restrict__ Wrh,
                      const float* __restrict__ Xz, const float* __restrict__ Xr,
                      float* __restrict__ zbuf, bf16_t* __restrict__ rhb, int t) {
  const int lane = threadIdx.x & 31;
  const int wave = threadIdx.x >> 5;
  const int r    = lane & 15;
  const int half = lane >> 4;
  const int mt   = wave & 1;
  const int nBase = blockIdx.x * 64 + (wave >> 1) * 16;

  v8f az = vzero8(), ar = vzero8();

  const bf16_t* Arow = hb + (size_t)(mt * 16 + r) * Hh;
  const bf16_t* Wzr  = Wzh + (size_t)(nBase + r) * Hh;
  const bf16_t* Wrr  = Wrh + (size_t)(nBase + r) * Hh;

  for (int k0 = 0; k0 < Hh; k0 += 32) {
    const v16bf a  = load_frag(Arow + k0, half);
    const v16bf bz = load_frag(Wzr + k0, half);
    const v16bf br = load_frag(Wrr + k0, half);
    az = __builtin_amdgcn_wmma_f32_16x16x32_bf16(false, a, false, bz, (short)0, az, false, false);
    ar = __builtin_amdgcn_wmma_f32_16x16x32_bf16(false, a, false, br, (short)0, ar, false, false);
  }

  const int n = nBase + r;
#pragma unroll
  for (int v = 0; v < 8; ++v) {
    const int b = mt * 16 + (half ? v + 8 : v);
    const size_t xi = ((size_t)b * Ss + t) * Hh + n;
    const size_t hi = (size_t)b * Hh + n;
    float z  = sigmoid_f(Xz[xi] + az[v]);
    float rr = sigmoid_f(Xr[xi] + ar[v]);
    zbuf[hi] = z;
    rhb[hi]  = (bf16_t)(rr * hf[hi]);
  }
}

// ---------------------------------------------------------------------------
// Step B: g = tanh(Xg[t] + (r*h) @ Wgh^T); h' = z*h + (1-z)*g
// writes hf/hb, next-layer bf16 activation y[t], hidden-state out at t==S-1.
// ---------------------------------------------------------------------------
__global__ __launch_bounds__(256)
void gru_stepB_kernel(const bf16_t* __restrict__ rhb, float* __restrict__ hf,
                      bf16_t* __restrict__ hb, const bf16_t* __restrict__ Wgh,
                      const float* __restrict__ Xg, const float* __restrict__ zbuf,
                      bf16_t* __restrict__ ybf, float* __restrict__ hid,
                      int layer, int t) {
  const int lane = threadIdx.x & 31;
  const int wave = threadIdx.x >> 5;
  const int r    = lane & 15;
  const int half = lane >> 4;
  const int mt   = wave & 1;
  const int nBase = blockIdx.x * 64 + (wave >> 1) * 16;

  v8f ag = vzero8();
  const bf16_t* Arow = rhb + (size_t)(mt * 16 + r) * Hh;
  const bf16_t* Wgr  = Wgh + (size_t)(nBase + r) * Hh;

  for (int k0 = 0; k0 < Hh; k0 += 32) {
    const v16bf a  = load_frag(Arow + k0, half);
    const v16bf bg = load_frag(Wgr + k0, half);
    ag = __builtin_amdgcn_wmma_f32_16x16x32_bf16(false, a, false, bg, (short)0, ag, false, false);
  }

  const int n = nBase + r;
#pragma unroll
  for (int v = 0; v < 8; ++v) {
    const int b = mt * 16 + (half ? v + 8 : v);
    const size_t xi = ((size_t)b * Ss + t) * Hh + n;
    const size_t hi = (size_t)b * Hh + n;
    float g  = tanh_f(Xg[xi] + ag[v]);
    float z  = zbuf[hi];
    float hn = z * hf[hi] + (1.0f - z) * g;
    hf[hi] = hn;
    hb[hi] = (bf16_t)hn;
    ybf[xi] = (bf16_t)hn;
    if (t == Ss - 1) hid[((size_t)b * Ll + layer) * Hh + n] = hn;
  }
}

// ---------------------------------------------------------------------------
// Host orchestration
// ---------------------------------------------------------------------------
extern "C" void kernel_launch(void* const* d_in, const int* in_sizes, int n_in,
                              void* d_out, int out_size, void* d_ws, size_t ws_size,
                              hipStream_t stream) {
  (void)in_sizes; (void)n_in; (void)out_size; (void)ws_size;

  const float* x   = (const float*)d_in[0];
  const float* h0  = (const float*)d_in[1];
  const float* Wzx = (const float*)d_in[2];
  const float* bz  = (const float*)d_in[3];
  const float* Wzh = (const float*)d_in[4];
  const float* Wrx = (const float*)d_in[5];
  const float* br  = (const float*)d_in[6];
  const float* Wrh = (const float*)d_in[7];
  const float* Wgx = (const float*)d_in[8];
  const float* bg  = (const float*)d_in[9];
  const float* Wgh = (const float*)d_in[10];
  const float* Wo  = (const float*)d_in[11];
  const float* bo  = (const float*)d_in[12];

  float* out_proj   = (float*)d_out;                         // [B,S,O]
  float* out_hidden = out_proj + (size_t)Bb * Ss * Oo;       // [B,L,H]

  // --- workspace carve ---
  char* ws = (char*)d_ws;
  size_t off = 0;
  auto carve = [&](size_t bytes) -> char* {
    char* p = ws + off;
    off += (bytes + 255) & ~(size_t)255;
    return p;
  };
  bf16_t* actA  = (bf16_t*)carve((size_t)MTOT * Hh * 2);
  bf16_t* actB  = (bf16_t*)carve((size_t)MTOT * Hh * 2);
  float*  Xz    = (float*)carve((size_t)MTOT * Hh * 4);
  float*  Xr    = (float*)carve((size_t)MTOT * Hh * 4);
  float*  Xg    = (float*)carve((size_t)MTOT * Hh * 4);
  bf16_t* Wzx_b = (bf16_t*)carve((size_t)Ll * Hh * Hh * 2);
  bf16_t* Wrx_b = (bf16_t*)carve((size_t)Ll * Hh * Hh * 2);
  bf16_t* Wgx_b = (bf16_t*)carve((size_t)Ll * Hh * Hh * 2);
  bf16_t* Wzh_b = (bf16_t*)carve((size_t)Ll * Hh * Hh * 2);
  bf16_t* Wrh_b = (bf16_t*)carve((size_t)Ll * Hh * Hh * 2);
  bf16_t* Wgh_b = (bf16_t*)carve((size_t)Ll * Hh * Hh * 2);
  bf16_t* Wo_b  = (bf16_t*)carve((size_t)Oo * Hh * 2);
  float*  hf    = (float*)carve((size_t)Bb * Hh * 4);
  bf16_t* hb    = (bf16_t*)carve((size_t)Bb * Hh * 2);
  float*  zbuf  = (float*)carve((size_t)Bb * Hh * 4);
  bf16_t* rhb   = (bf16_t*)carve((size_t)Bb * Hh * 2);

  auto cvt = [&](const float* in, bf16_t* out, size_t n) {
    unsigned blocks = (unsigned)((n + 255) / 256);
    cvt_f32_bf16<<<blocks, 256, 0, stream>>>(in, out, n);
  };

  // --- convert inputs / weights to bf16 ---
  cvt(x,   actA,  (size_t)MTOT * Hh);
  cvt(Wzx, Wzx_b, (size_t)Ll * Hh * Hh);
  cvt(Wrx, Wrx_b, (size_t)Ll * Hh * Hh);
  cvt(Wgx, Wgx_b, (size_t)Ll * Hh * Hh);
  cvt(Wzh, Wzh_b, (size_t)Ll * Hh * Hh);
  cvt(Wrh, Wrh_b, (size_t)Ll * Hh * Hh);
  cvt(Wgh, Wgh_b, (size_t)Ll * Hh * Hh);
  cvt(Wo,  Wo_b,  (size_t)Oo * Hh);

  const dim3 gemm_grid(Hh / 64, MTOT / 128);   // (16, 128)
  const dim3 gemm_blk(256);

  bf16_t* cur = actA;
  bf16_t* nxt = actB;
  for (int l = 0; l < Ll; ++l) {
    const size_t wo = (size_t)l * Hh * Hh;

    // input-side gate projections as big WMMA GEMMs
    gemm_bias_kernel<<<gemm_grid, gemm_blk, 0, stream>>>(cur, Wzx_b + wo, bz + (size_t)l * Hh, Xz, MTOT, Hh, Hh);
    gemm_bias_kernel<<<gemm_grid, gemm_blk, 0, stream>>>(cur, Wrx_b + wo, br + (size_t)l * Hh, Xr, MTOT, Hh, Hh);
    gemm_bias_kernel<<<gemm_grid, gemm_blk, 0, stream>>>(cur, Wgx_b + wo, bg + (size_t)l * Hh, Xg, MTOT, Hh, Hh);

    init_h_kernel<<<(Bb * Hh + 255) / 256, 256, 0, stream>>>(h0, l, hf, hb);

    // sequential scan; recurrent weights (6 MB/layer bf16) stay hot in 192 MB L2
    for (int t = 0; t < Ss; ++t) {
      gru_stepA_kernel<<<16, 256, 0, stream>>>(hb, hf, Wzh_b + wo, Wrh_b + wo,
                                               Xz, Xr, zbuf, rhb, t);
      gru_stepB_kernel<<<16, 256, 0, stream>>>(rhb, hf, hb, Wgh_b + wo,
                                               Xg, zbuf, nxt, out_hidden, l, t);
    }
    bf16_t* tmp = cur; cur = nxt; nxt = tmp;
  }

  // output projection straight into d_out
  gemm_bias_kernel<<<dim3(Oo / 64, MTOT / 128), gemm_blk, 0, stream>>>(
      cur, Wo_b, bo, out_proj, MTOT, Oo, Hh);
}
